// VEConv_83743272337869
// MI455X (gfx1250) — matrix operand order repaired and checked by reading, not compiled, not run on hardware
//
#include <hip/hip_runtime.h>

typedef __attribute__((ext_vector_type(8)))  __bf16 v8bf;
typedef __attribute__((ext_vector_type(16))) __bf16 v16bf;
typedef __attribute__((ext_vector_type(8)))  float  v8f;

#define EDGES_PER_WAVE  16
#define WAVES_PER_BLOCK 8
#define EDGES_PER_BLOCK (EDGES_PER_WAVE * WAVES_PER_BLOCK)

// shifted softplus: (1/0.5)*log(1+exp(0.5*x)), linear when 0.5*x > 14.
// Use hardware v_exp_f32 / v_log_f32 (via __expf/__logf) instead of the
// libm log1pf expansion: argument of the log is in [1, ~1.2e6] (hx <= 14),
// so there is no cancellation and fast-math intrinsics are plenty accurate.
__device__ __forceinline__ float ssp(float x) {
    float hx = 0.5f * x;
    float e  = __expf(hx);
    float sp = 2.0f * __logf(1.0f + e);
    return (hx > 14.0f) ? x : sp;
}

__device__ __forceinline__ v8bf cvt8(float4 a, float4 b) {
    v8bf r;
    r[0] = (__bf16)a.x; r[1] = (__bf16)a.y; r[2] = (__bf16)a.z; r[3] = (__bf16)a.w;
    r[4] = (__bf16)b.x; r[5] = (__bf16)b.y; r[6] = (__bf16)b.z; r[7] = (__bf16)b.w;
    return r;
}

__device__ __forceinline__ v16bf cat16(v8bf lo, v8bf hi) {
    return __builtin_shufflevector(lo, hi, 0,1,2,3,4,5,6,7,8,9,10,11,12,13,14,15);
}

// A-fragment (16x32 bf16) from a global f32 row.
// Per ISA layout: lane<16 holds K = klo..klo+7 (elems 0..7) and klo+16..klo+23
// (elems 8..15); lanes>=16 are offset by +8 (folded into klo by the caller).
__device__ __forceinline__ v16bf load_a_global(const float* __restrict__ row, int klo) {
    float4 a0 = *(const float4*)(row + klo);
    float4 a1 = *(const float4*)(row + klo + 4);
    float4 b0 = *(const float4*)(row + klo + 16);
    float4 b1 = *(const float4*)(row + klo + 20);
    return cat16(cvt8(a0, a1), cvt8(b0, b1));
}

__device__ __forceinline__ v16bf load_a_lds(const __bf16* row, int klo) {
    v8bf lo = *(const v8bf*)(row + klo);
    v8bf hi = *(const v8bf*)(row + klo + 16);
    return cat16(lo, hi);
}

__global__ void __launch_bounds__(256)
veconv_zero_kernel(float4* __restrict__ out, int n4) {
    int i = blockIdx.x * blockDim.x + threadIdx.x;
    if (i < n4) out[i] = make_float4(0.f, 0.f, 0.f, 0.f);
}

__global__ void __launch_bounds__(256)
veconv_kernel(const float* __restrict__ rbf,
              const float* __restrict__ edge_f,
              const float* __restrict__ node_feat,
              const int*   __restrict__ src,
              const int*   __restrict__ dst,
              const float* __restrict__ W1, const float* __restrict__ b1,
              const float* __restrict__ W2, const float* __restrict__ b2,
              const float* __restrict__ W3, const float* __restrict__ b3,
              float* __restrict__ out,
              int n_edges)
{
    // Weights staged transposed ([N][K]) in bf16 so B-fragments are contiguous.
    __shared__ __attribute__((aligned(32))) __bf16 WT1[64 * 128];
    __shared__ __attribute__((aligned(32))) __bf16 WT2[64 * 64];
    __shared__ __attribute__((aligned(32))) __bf16 WT3[64 * 64];
    __shared__ __attribute__((aligned(32))) __bf16 hbuf[WAVES_PER_BLOCK * 16 * 64];

    const int tid = threadIdx.x;
    for (int i = tid; i < 128 * 64; i += 256) {
        int k = i >> 6, n = i & 63;
        WT1[n * 128 + k] = (__bf16)W1[i];
    }
    for (int i = tid; i < 64 * 64; i += 256) {
        int k = i >> 6, n = i & 63;
        WT2[n * 64 + k] = (__bf16)W2[i];
        WT3[n * 64 + k] = (__bf16)W3[i];
    }
    __syncthreads();

    const int wid  = tid >> 5;
    const int lane = tid & 31;
    const int l15  = lane & 15;
    const int kg   = lane >> 4;         // K-group (which half of the 32-K step)
    const long e0  = (long)blockIdx.x * EDGES_PER_BLOCK + (long)wid * EDGES_PER_WAVE;
    if (e0 + EDGES_PER_WAVE > n_edges) return;   // E is a multiple of 128 here

    // ---------------- GEMM1: h1 = rbf[16x128] @ W1[128x64] ----------------
    const float* rbf_row = rbf + (e0 + l15) * 128;
    v8f acc1[4] = {};
    #pragma unroll
    for (int kk = 0; kk < 4; kk++) {
        v16bf a = load_a_global(rbf_row, kk * 32 + kg * 8);
        #pragma unroll
        for (int t = 0; t < 4; t++) {
            v16bf b = load_a_lds(WT1 + (t * 16 + l15) * 128, kk * 32 + kg * 8);
            acc1[t] = __builtin_amdgcn_wmma_f32_16x16x32_bf16(
                false, a, false, b, (short)0, acc1[t], false, false);
        }
    }

    // bias + shifted softplus -> per-wave LDS h buffer (bf16, [M][K] layout)
    __bf16* hb = hbuf + wid * (16 * 64);
    #pragma unroll
    for (int t = 0; t < 4; t++) {
        float bb = b1[t * 16 + l15];
        #pragma unroll
        for (int v = 0; v < 8; v++) {
            // C/D layout: element v on this lane is M = v + 8*kg, N = t*16+l15
            float x = ssp(acc1[t][v] + bb);
            hb[(v + 8 * kg) * 64 + t * 16 + l15] = (__bf16)x;
        }
    }

    // ---------------- GEMM2: h2 = h[16x64] @ W2[64x64] ----------------
    const __bf16* hrow = hb + l15 * 64;
    v8f acc2[4] = {};
    #pragma unroll
    for (int kk = 0; kk < 2; kk++) {
        v16bf a = load_a_lds(hrow, kk * 32 + kg * 8);
        #pragma unroll
        for (int t = 0; t < 4; t++) {
            v16bf b = load_a_lds(WT2 + (t * 16 + l15) * 64, kk * 32 + kg * 8);
            acc2[t] = __builtin_amdgcn_wmma_f32_16x16x32_bf16(
                false, a, false, b, (short)0, acc2[t], false, false);
        }
    }

    // ---------------- GEMM3: edge_new = edge_f[16x64] @ W3[64x64] ----------------
    const float* ef_row = edge_f + (e0 + l15) * 64;
    v8f acc3[4] = {};
    #pragma unroll
    for (int kk = 0; kk < 2; kk++) {
        v16bf a = load_a_global(ef_row, kk * 32 + kg * 8);
        #pragma unroll
        for (int t = 0; t < 4; t++) {
            v16bf b = load_a_lds(WT3 + (t * 16 + l15) * 64, kk * 32 + kg * 8);
            acc3[t] = __builtin_amdgcn_wmma_f32_16x16x32_bf16(
                false, a, false, b, (short)0, acc3[t], false, false);
        }
    }

    // ---------------- fuse: msg = node_feat[src]*h2 + edge_new; scatter-add ----------------
    int se[8], de[8];
    #pragma unroll
    for (int v = 0; v < 8; v++) {
        long e = e0 + 8 * kg + v;
        se[v] = src[e];
        de[v] = dst[e];
    }
    #pragma unroll
    for (int t = 0; t < 4; t++) {
        int   N   = t * 16 + l15;
        float b2v = b2[N];
        float b3v = b3[N];
        #pragma unroll
        for (int v = 0; v < 8; v++) {
            float h2 = acc2[t][v] + b2v;
            float en = acc3[t][v] + b3v;
            float nf = node_feat[(long)se[v] * 64 + N];
            float m  = fmaf(nf, h2, en);
            __hip_atomic_fetch_add(out + (long)de[v] * 64 + N, m,
                                   __ATOMIC_RELAXED, __HIP_MEMORY_SCOPE_AGENT);
        }
    }
}

extern "C" void kernel_launch(void* const* d_in, const int* in_sizes, int n_in,
                              void* d_out, int out_size, void* d_ws, size_t ws_size,
                              hipStream_t stream) {
    const float* rbf       = (const float*)d_in[0];
    const float* edge_f    = (const float*)d_in[1];
    const float* node_feat = (const float*)d_in[2];
    const int*   src       = (const int*)d_in[3];
    const int*   dst       = (const int*)d_in[4];
    const float* W1        = (const float*)d_in[5];
    const float* b1        = (const float*)d_in[6];
    const float* W2        = (const float*)d_in[7];
    const float* b2        = (const float*)d_in[8];
    const float* W3        = (const float*)d_in[9];
    const float* b3        = (const float*)d_in[10];
    float*       out       = (float*)d_out;

    const int E  = in_sizes[3];          // number of edges (src length)
    const int n4 = out_size / 4;         // out_size is N_NODES*64, divisible by 4

    veconv_zero_kernel<<<(n4 + 255) / 256, 256, 0, stream>>>((float4*)out, n4);

    const int blocks = (E + EDGES_PER_BLOCK - 1) / EDGES_PER_BLOCK;
    veconv_kernel<<<blocks, 256, 0, stream>>>(rbf, edge_f, node_feat, src, dst,
                                              W1, b1, W2, b2, W3, b3, out, E);
}
